// SelfAttention_30425548324945
// MI455X (gfx1250) — compile-verified, matrix operand order checked by reading, not run on hardware
//
#include <hip/hip_runtime.h>
#include <hip/hip_bf16.h>
#include <stdint.h>

// Problem dims (fixed by reference)
#define B_  2
#define S_  2048
#define H_  1024
#define NH_ 16
#define HD_ 64

// padded LDS tile row stride (halves): 72 -> 36 dwords/row, conflict-free rows
#define TPAD 72

typedef __attribute__((ext_vector_type(16))) short     v16s;
typedef __attribute__((ext_vector_type(16))) __bf16    v16bf;
typedef __attribute__((ext_vector_type(2)))  __bf16    v2bf;
typedef __attribute__((ext_vector_type(8)))  float     v8f;
typedef __attribute__((ext_vector_type(4)))  unsigned  v4u;
typedef __attribute__((ext_vector_type(8)))  unsigned  v8u;
typedef __attribute__((ext_vector_type(2)))  unsigned  v2u;

// Hardware f32->bf16 (backend packs pairs into v_cvt_pk_bf16_f32)
__device__ __forceinline__ unsigned short f2bf(float f) {
  __bf16 h = (__bf16)f;
  return __builtin_bit_cast(unsigned short, h);
}

// Two f32 -> one packed dword of bf16 (lo = a, hi = b)
__device__ __forceinline__ unsigned pack_bf16(float a, float b) {
  v2bf v; v[0] = (__bf16)a; v[1] = (__bf16)b;
  return __builtin_bit_cast(unsigned, v);
}

__device__ __forceinline__ v8f zero8() {
  v8f z;
#pragma unroll
  for (int i = 0; i < 8; ++i) z[i] = 0.0f;
  return z;
}

// ---- CDNA5 async global->LDS copy (ASYNCcnt-tracked), ISA 15.18.3 op 98 ----
// lds_off: 32-bit LDS byte address (low 32 bits of a generic shared pointer)
__device__ __forceinline__ void async_load_b128(unsigned lds_off, const void* gptr) {
  asm volatile("global_load_async_to_lds_b128 %0, %1, off"
               :: "v"(lds_off), "v"(gptr) : "memory");
}
__device__ __forceinline__ void wait_asynccnt0() {
  asm volatile("s_wait_asynccnt 0x0" ::: "memory");
}

// A/B fragment (16x32 bf16) from a row-major bf16 source.
// p already points at this lane's row start + k0 + kb (kb = (lane>=16)?8:0).
__device__ __forceinline__ v16s load_frag_b(const unsigned short* p) {
  v4u a = *(const v4u*)p;
  v4u b = *(const v4u*)(p + 16);
  v8u t;
#pragma unroll
  for (int i = 0; i < 4; ++i) { t[i] = a[i]; t[i + 4] = b[i]; }
  return __builtin_bit_cast(v16s, t);
}

// Same fragment, converting from f32 source (global or LDS), dword-packed.
__device__ __forceinline__ v16s cvt_frag_f32(const float* p) {
  v8u t;
#pragma unroll
  for (int i = 0; i < 4; ++i) {
    t[i]     = pack_bf16(p[2 * i],      p[2 * i + 1]);
    t[i + 4] = pack_bf16(p[16 + 2 * i], p[16 + 2 * i + 1]);
  }
  return __builtin_bit_cast(v16s, t);
}

__device__ __forceinline__ v8f wmma_bf16(v16s a, v16s b, v8f c) {
  return __builtin_amdgcn_wmma_f32_16x16x32_bf16(
      false, __builtin_bit_cast(v16bf, a),
      false, __builtin_bit_cast(v16bf, b),
      (short)0, c, false, false);
}

// ---------------------------------------------------------------------------
// Kernel 0: convert Wo (H x H f32) -> bf16, 4 elements/thread, packed stores.
// ---------------------------------------------------------------------------
__global__ __launch_bounds__(256) void k_conv_wo(const float* __restrict__ Wo,
                                                 unsigned short* __restrict__ Wob) {
  int i = (blockIdx.x * 256 + threadIdx.x) * 4;
  if (i < H_ * H_) {
    const float* p = Wo + i;
    v2u d;
    d[0] = pack_bf16(p[0], p[1]);
    d[1] = pack_bf16(p[2], p[3]);
    *(v2u*)(Wob + i) = d;
  }
}

// ---------------------------------------------------------------------------
// Kernel 1: QKV projection (bf16 WMMA, K=64).
// X = hidden viewed as [B*S*NH, 64] f32 (contiguous).
// Outputs (bf16): qb,kb as [head][S][64]; v stored TRANSPOSED vT[head][64][S].
// w loop NOT unrolled (pressure); W/dst/base picked via ternaries on the
// uniform index (no memory-backed pointer array -> keeps global_load path).
// ---------------------------------------------------------------------------
__global__ __launch_bounds__(128) void k_qkv(const float* __restrict__ X,
                                             const float* __restrict__ Wq,
                                             const float* __restrict__ Wk,
                                             const float* __restrict__ Wv,
                                             unsigned short* __restrict__ qb,
                                             unsigned short* __restrict__ kb,
                                             unsigned short* __restrict__ vT) {
  const int wave = threadIdx.x >> 5;
  const int lane = threadIdx.x & 31;
  const int r    = lane & 15;
  const int kbo  = (lane >> 4) * 8;
  const int row0 = blockIdx.x * 64 + wave * 16;      // 16-aligned -> nh=0 at m=0
  const int mhi  = (lane >> 4) * 8;

  const int bs = row0 >> 4;                          // b*S + s (same for all 16 rows)
  const int s  = bs & (S_ - 1);
  const int b  = bs >> 11;

  const float* xr = X + (size_t)(row0 + r) * 64;
  v16s ax0 = cvt_frag_f32(xr + 0  + kbo);
  v16s ax1 = cvt_frag_f32(xr + 32 + kbo);

  const size_t qkstride = (size_t)S_ * HD_;          // per-head step (q/k/vT alike)

#pragma unroll 1
  for (int w = 0; w < 3; ++w) {
    const float* W = (w == 0) ? Wq : ((w == 1) ? Wk : Wv);
    unsigned short* dst = (w == 0) ? qb : ((w == 1) ? kb : vT);
#pragma unroll
    for (int nt = 0; nt < 4; ++nt) {
      const float* wr = W + (size_t)(nt * 16 + r) * 64;
      v16s b0 = cvt_frag_f32(wr + 0  + kbo);
      v16s b1 = cvt_frag_f32(wr + 32 + kbo);
      v8f acc = zero8();
      acc = wmma_bf16(ax0, b0, acc);
      acc = wmma_bf16(ax1, b1, acc);
      const int col = nt * 16 + r;

      // convert first (8 distinct regs), then store (backend can clause)
      unsigned short bfv[8];
#pragma unroll
      for (int i = 0; i < 8; ++i) bfv[i] = f2bf(acc[i]);

      const size_t qkbase = ((size_t)b * NH_ * S_ + s) * HD_ + col;   // q/k layout
      const size_t vbase  = ((size_t)b * NH_ * HD_ + col) * S_ + s;   // vT layout
      const size_t base   = (w == 2) ? vbase : qkbase;
#pragma unroll
      for (int i = 0; i < 8; ++i) {
        dst[base + (size_t)(i + mhi) * qkstride] = bfv[i];            // head step
      }
    }
  }
}

// ---------------------------------------------------------------------------
// Kernel 2: fused scores + ctx, per (head, 64-row q tile).
// Double-buffered async staging of k / vT tiles into LDS
// (global_load_async_to_lds_b128 + s_wait_asynccnt), 16 WMMA / wave / tile.
// ---------------------------------------------------------------------------
__device__ __forceinline__ void stage_tiles(const unsigned short* kh,
                                            const unsigned short* vh,
                                            int j0,
                                            unsigned short* ktb,
                                            unsigned short* vtb,
                                            int tid) {
  // k tile: 64 rows x 128B, contiguous 8KB in global; LDS rows padded to 144B
  {
    const char* g = (const char*)(kh + (size_t)j0 * HD_);
    unsigned lb = (unsigned)(uintptr_t)ktb;
#pragma unroll
    for (int c = 0; c < 4; ++c) {
      int o   = tid * 16 + c * 2048;           // global byte offset 0..8191
      int row = o >> 7;
      int col = o & 127;
      async_load_b128(lb + (unsigned)(row * (TPAD * 2) + col), g + o);
    }
  }
  // vT tile: 64 rows x 128B, global row stride S_*2 bytes
  {
    int row = tid >> 1;
    int off = (tid & 1) * 64;
    const char* g = (const char*)(vh + (size_t)row * S_ + j0) + off;
    unsigned lb = (unsigned)(uintptr_t)vtb + (unsigned)(row * (TPAD * 2) + off);
#pragma unroll
    for (int c = 0; c < 4; ++c)
      async_load_b128(lb + c * 16, g + c * 16);
  }
}

__global__ __launch_bounds__(128) void k_attn(const unsigned short* __restrict__ qb,
                                              const unsigned short* __restrict__ kb,
                                              const unsigned short* __restrict__ vT,
                                              const float* __restrict__ mask,
                                              float* __restrict__ scores,
                                              unsigned short* __restrict__ ctxb) {
  __shared__ float ps[4][16][68];                    // per-wave 16x64 (stride 68)
  __shared__ unsigned short kt[2][64 * TPAD];        // double-buffered k tile
  __shared__ unsigned short vt[2][64 * TPAD];        // double-buffered vT tile

  const int tid  = threadIdx.x;
  const int wave = tid >> 5;
  const int lane = tid & 31;
  const int r    = lane & 15;
  const int kbo  = (lane >> 4) * 8;
  const int mhi  = (lane >> 4) * 8;

  const int head = blockIdx.y;                       // 0..31
  const int b    = head >> 4;
  const int nh   = head & 15;
  const int m0   = blockIdx.x * 64 + wave * 16;      // this wave's q rows

  const unsigned short* qh = qb + (size_t)head * S_ * HD_;
  const unsigned short* kh = kb + (size_t)head * S_ * HD_;
  const unsigned short* vh = vT + (size_t)head * HD_ * S_;

  // Persistent A fragments: q rows m0..m0+15, K = 0..63
  const unsigned short* qr = qh + (size_t)(m0 + r) * HD_;
  v16s aq0 = load_frag_b(qr + 0  + kbo);
  v16s aq1 = load_frag_b(qr + 32 + kbo);

  v8f cacc[4];
#pragma unroll
  for (int i = 0; i < 4; ++i) cacc[i] = zero8();

  float (*myps)[68] = ps[wave];

  // scores row-store mapping: each lane stores half a row (32 f32 = 128B)
  const int srow = lane >> 1;
  const int scol = (lane & 1) * 32;

  // preload first tile pair
  stage_tiles(kh, vh, 0, kt[0], vt[0], tid);

  for (int j0 = 0; j0 < S_; j0 += 64) {
    const int buf = (j0 >> 6) & 1;
    wait_asynccnt0();          // this wave's staged tiles have landed in LDS
    __syncthreads();           // ... and everyone else's too

    if (j0 + 64 < S_)          // fire next tile while we compute this one
      stage_tiles(kh, vh, j0 + 64, kt[buf ^ 1], vt[buf ^ 1], tid);

    const unsigned short* ktb = kt[buf];
    const unsigned short* vtb = vt[buf];

    // ---- scores tile: 16 x 64 -> LDS (scaled + mask applied) ----
#pragma unroll
    for (int jt = 0; jt < 4; ++jt) {
      const unsigned short* kr = ktb + (jt * 16 + r) * TPAD;
      v16s bk0 = load_frag_b(kr + 0  + kbo);
      v16s bk1 = load_frag_b(kr + 32 + kbo);
      v8f s = zero8();
      s = wmma_bf16(aq0, bk0, s);
      s = wmma_bf16(aq1, bk1, s);
      const float mv = mask[b * S_ + j0 + jt * 16 + r];
#pragma unroll
      for (int i = 0; i < 8; ++i) {
        myps[i + mhi][jt * 16 + r] = s[i] * 0.125f + mv;
      }
    }

    // ---- wide coalesced scores store: 8 x global_store_b128 per lane ----
    {
      const float* srcp = &myps[srow][scol];
      float* dstp = &scores[((size_t)head * S_ + (m0 + srow)) * S_ + j0 + scol];
#pragma unroll
      for (int i = 0; i < 8; ++i) {
        ((v4u*)dstp)[i] = ((const v4u*)srcp)[i];
      }
    }

    // ---- P (bf16) fragments from LDS: rows r, K = local j 0..63 ----
    v16s pf0 = cvt_frag_f32(&myps[r][0  + kbo]);
    v16s pf1 = cvt_frag_f32(&myps[r][32 + kbo]);

    // ---- ctx += P @ v : B fragments from LDS vT tile ----
#pragma unroll
    for (int nt = 0; nt < 4; ++nt) {
      const unsigned short* vr = vtb + (nt * 16 + r) * TPAD;
      v16s bv0 = load_frag_b(vr + 0  + kbo);
      v16s bv1 = load_frag_b(vr + 32 + kbo);
      cacc[nt] = wmma_bf16(pf0, bv0, cacc[nt]);
      cacc[nt] = wmma_bf16(pf1, bv1, cacc[nt]);
    }
    __syncthreads();           // protect ps + tile buffer reuse
  }

  // ---- write ctx (bf16) to ws as [b*S+s][H]: convert all, then store ----
#pragma unroll
  for (int nt = 0; nt < 4; ++nt) {
    const int h = nh * HD_ + nt * 16 + r;
    unsigned short bfv[8];
#pragma unroll
    for (int i = 0; i < 8; ++i) bfv[i] = f2bf(cacc[nt][i]);
    const size_t base = ((size_t)b * S_ + (m0 + mhi)) * H_ + h;
#pragma unroll
    for (int i = 0; i < 8; ++i) {
      ctxb[base + (size_t)i * H_] = bfv[i];
    }
  }
}

// ---------------------------------------------------------------------------
// Kernel 3: out = ctx @ Wo^T.  ctxb [B*S, H] bf16, Wob [H,H] bf16.
// Workgroup: 64 rows x 128 cols; wave: 16 rows x 128 cols; K=1024.
// ---------------------------------------------------------------------------
__global__ __launch_bounds__(128) void k_proj(const unsigned short* __restrict__ ctxb,
                                              const unsigned short* __restrict__ Wob,
                                              float* __restrict__ out) {
  const int wave = threadIdx.x >> 5;
  const int lane = threadIdx.x & 31;
  const int r    = lane & 15;
  const int kbo  = (lane >> 4) * 8;
  const int mhi  = (lane >> 4) * 8;

  const int m0 = blockIdx.x * 64 + wave * 16;        // row in [0, B*S)
  const int n0 = blockIdx.y * 128;

  v8f acc[8];
#pragma unroll
  for (int i = 0; i < 8; ++i) acc[i] = zero8();

  const unsigned short* arow = ctxb + (size_t)(m0 + r) * H_;

  for (int k0 = 0; k0 < H_; k0 += 32) {
    v16s a = load_frag_b(arow + k0 + kbo);
#pragma unroll
    for (int nt = 0; nt < 8; ++nt) {
      const unsigned short* wr = Wob + (size_t)(n0 + nt * 16 + r) * H_ + k0 + kbo;
      v16s bfr = load_frag_b(wr);
      acc[nt] = wmma_bf16(a, bfr, acc[nt]);
    }
  }

#pragma unroll
  for (int nt = 0; nt < 8; ++nt) {
    const int col = n0 + nt * 16 + r;
    const size_t base = (size_t)(m0 + mhi) * H_ + col;
#pragma unroll
    for (int i = 0; i < 8; ++i) {
      out[base + (size_t)i * H_] = acc[nt][i];
    }
  }
}

// ---------------------------------------------------------------------------
extern "C" void kernel_launch(void* const* d_in, const int* in_sizes, int n_in,
                              void* d_out, int out_size, void* d_ws, size_t ws_size,
                              hipStream_t stream) {
  const float* hidden = (const float*)d_in[0];
  const float* mask   = (const float*)d_in[1];
  const float* Wq     = (const float*)d_in[2];
  const float* Wk     = (const float*)d_in[3];
  const float* Wv     = (const float*)d_in[4];
  const float* Wo     = (const float*)d_in[5];

  float* out    = (float*)d_out;
  float* scores = out + (size_t)B_ * S_ * H_;        // tuple: (out, scores)

  char* ws = (char*)d_ws;
  unsigned short* qb   = (unsigned short*)(ws);                        // 8 MB
  unsigned short* kb   = (unsigned short*)(ws + (size_t)8  * 1048576); // 8 MB
  unsigned short* vT   = (unsigned short*)(ws + (size_t)16 * 1048576); // 8 MB
  unsigned short* ctxb = (unsigned short*)(ws + (size_t)24 * 1048576); // 8 MB
  unsigned short* Wob  = (unsigned short*)(ws + (size_t)32 * 1048576); // 2 MB

  k_conv_wo<<<(H_ * H_) / 1024, 256, 0, stream>>>(Wo, Wob);
  k_qkv<<<(B_ * S_ * NH_) / 64, 128, 0, stream>>>(hidden, Wq, Wk, Wv, qb, kb, vT);
  k_attn<<<dim3(S_ / 64, B_ * NH_), 128, 0, stream>>>(qb, kb, vT, mask, scores, ctxb);
  k_proj<<<dim3((B_ * S_) / 64, H_ / 128), 128, 0, stream>>>(ctxb, Wob, out);
}
